// PhaseAwareJDKANMixingBlock_56813827391732
// MI455X (gfx1250) — matrix operand order repaired and checked by reading, not compiled, not run on hardware
//
#include <hip/hip_runtime.h>
#include <hip/hip_bf16.h>
#include <math.h>

// ---------------- problem constants ----------------
#define BB 16
#define TT 1024
#define CC 512
#define KK 25
#define NW 4
#define MROWS (BB * TT)          // 16384
#define KCHEB (CC * 4)           // 2048
#define KWAVE (NW * CC)          // 2048

typedef __attribute__((ext_vector_type(16))) __bf16 v16bf;
typedef __attribute__((ext_vector_type(8)))  float  v8f;

__device__ __forceinline__ unsigned short f2bf(float f) {
    unsigned u = __float_as_uint(f);
    unsigned r = u + 0x7FFFu + ((u >> 16) & 1u);   // RNE
    return (unsigned short)(r >> 16);
}
__device__ __forceinline__ float bf2f(unsigned short s) {
    return __uint_as_float(((unsigned)s) << 16);
}
// branch-free transcendentals on v_exp_f32 / v_rcp_f32 (TRANS pipe,
// co-executes with WMMA; bf16 inputs don't need IEEE-rounded division)
__device__ __forceinline__ float fast_rcp(float x) {
    return __builtin_amdgcn_rcpf(x);
}
__device__ __forceinline__ float fast_tanh(float x) {
    float e = __expf(2.0f * x);
    return 1.0f - 2.0f * fast_rcp(e + 1.0f);   // ±1 saturating, no NaN at |x| large
}
__device__ __forceinline__ float fast_sigmoid(float x) {
    return fast_rcp(1.0f + __expf(-x));
}
// unpack 16 consecutive bf16 via two 128-bit global loads
__device__ __forceinline__ void load16bf(const unsigned short* __restrict__ p,
                                         float* __restrict__ out) {
    const uint4* q = (const uint4*)p;
    uint4 a = q[0], b = q[1];
    unsigned w[8] = {a.x, a.y, a.z, a.w, b.x, b.y, b.z, b.w};
    #pragma unroll
    for (int i = 0; i < 8; ++i) {
        out[2 * i]     = bf2f((unsigned short)(w[i] & 0xffffu));
        out[2 * i + 1] = bf2f((unsigned short)(w[i] >> 16));
    }
}

// ============================================================
// Weight packing: float32 -> bf16, PRE-PERMUTED into WMMA
// B-fragment order:  Wp[kTile][nTile][lane][e]
//   lane = (kk>>4)*16 + (n&15),  e = kk&15,  kk = k&31
// so each (32-K x 128-N) GEMM tile is one contiguous 8KB block
// consumable by a straight async global->LDS copy.
//   MODE 0: cheby_coeffs[c,o,d], k = c*4+d,  n = o
//   MODE 1: wavelet_w[w,c,o],    k = w*512+c, n = o
//   MODE 2: amp_w[k,o],          n = o
// ============================================================
template <int MODE>
__global__ void pack_weight_kernel(const float* __restrict__ src,
                                   unsigned short* __restrict__ dst) {
    int idx   = blockIdx.x * 256 + threadIdx.x;
    int e     = idx & 15;
    int lane  = (idx >> 4) & 31;
    int blk   = idx >> 9;
    int kTile = blk >> 5;                 // 32 nTiles per kTile (N=512)
    int nTile = blk & 31;
    int kk    = ((lane >> 4) << 4) + e;
    int k     = kTile * 32 + kk;
    int n     = nTile * 16 + (lane & 15);
    float v;
    if (MODE == 0)      { int c = k >> 2, d = k & 3;   v = src[(c * CC + n) * 4 + d]; }
    else if (MODE == 1) { int w = k >> 9, c = k & 511; v = src[(w * CC + c) * CC + n]; }
    else                {                              v = src[k * CC + n]; }
    dst[idx] = f2bf(v);
}

// ============================================================
// Decomposition scan: causal MA(K=25), spike, dwconv3 x3,
// time-pooling. One thread per (b,c) column, sequential over T.
// ============================================================
__global__ void decomp_scan_kernel(const float* __restrict__ x,
                                   const float* __restrict__ tw, const float* __restrict__ tb_,
                                   const float* __restrict__ sw, const float* __restrict__ sb_,
                                   const float* __restrict__ lw, const float* __restrict__ lb_,
                                   unsigned short* __restrict__ trend_ctx_bf,
                                   unsigned short* __restrict__ spike_ctx_bf,
                                   float* __restrict__ slope_pool,
                                   float* __restrict__ value_pool) {
    int gid = blockIdx.x * 256 + threadIdx.x;          // < B*C = 8192
    int b = gid >> 9;
    int c = gid & 511;
    const float tw0 = tw[c * 3 + 0], tw1 = tw[c * 3 + 1], tw2 = tw[c * 3 + 2], tbv = tb_[c];
    const float sw0 = sw[c * 3 + 0], sw1 = sw[c * 3 + 1], sw2 = sw[c * 3 + 2], sbv = sb_[c];
    const float lw0 = lw[c * 3 + 0], lw1 = lw[c * 3 + 1], lw2 = lw[c * 3 + 2], lbv = lb_[c];

    const size_t base = (size_t)(b * TT) * CC + c;
    float x0 = x[base];
    float runsum = (float)KK * x0;                     // window fully clamped at t=0
    float tr_cur = x0, sp_cur = 0.0f;
    float tr_prev = tr_cur, sp_prev = sp_cur;          // replicate pad at t=-1
    float slope_sum = 0.0f, val_sum = 0.0f;

    for (int t = 0; t < TT; ++t) {
        float tr_next, sp_next;
        if (t + 1 < TT) {
            float xn = x[base + (size_t)(t + 1) * CC];
            int rem = t + 1 - KK; if (rem < 0) rem = 0;
            float xr = x[base + (size_t)rem * CC];
            runsum += xn - xr;
            tr_next = runsum * (1.0f / (float)KK);
            sp_next = xn - tr_next;
        } else {                                        // replicate pad at t=T
            tr_next = tr_cur; sp_next = sp_cur;
        }
        float tctx = tr_prev * tw0 + tr_cur * tw1 + tr_next * tw2 + tbv;
        float sctx = sp_prev * sw0 + sp_cur * sw1 + sp_next * sw2 + sbv;
        float slp  = tr_prev * lw0 + tr_cur * lw1 + tr_next * lw2 + lbv;
        slope_sum += slp;
        val_sum   += tr_cur;
        size_t row = (size_t)(b * TT + t);
        trend_ctx_bf[row * CC + c] = f2bf(tctx);
        spike_ctx_bf[row * CC + c] = f2bf(sctx);
        tr_prev = tr_cur; sp_prev = sp_cur;
        tr_cur = tr_next; sp_cur = sp_next;
    }
    slope_pool[b * CC + c] = slope_sum * (1.0f / (float)TT);
    value_pool[b * CC + c] = val_sum   * (1.0f / (float)TT);
}

// ============================================================
// Hypernet layer 1: h = silu([slope_pool, value_pool] @ W1 + b1)
// ============================================================
__global__ void hyper1_kernel(const float* __restrict__ slope_pool,
                              const float* __restrict__ value_pool,
                              const float* __restrict__ w1, const float* __restrict__ b1,
                              float* __restrict__ h) {
    int gid = blockIdx.x * 256 + threadIdx.x;          // < B*C = 8192
    int b = gid >> 9, j = gid & 511;
    float acc = b1[j];
    for (int i = 0; i < CC; ++i) acc += slope_pool[b * CC + i] * w1[i * CC + j];
    for (int i = 0; i < CC; ++i) acc += value_pool[b * CC + i] * w1[(CC + i) * CC + j];
    h[gid] = acc * fast_sigmoid(acc);                  // SiLU
}

// ============================================================
// Hypernet layer 2 + Ricker wavelet bank psi[b,t,w]
// ============================================================
__global__ void hyper2_psi_kernel(const float* __restrict__ h,
                                  const float* __restrict__ w2, const float* __restrict__ b2,
                                  float* __restrict__ psi) {
    __shared__ float s_a[NW];
    __shared__ float s_b[NW];
    int b = blockIdx.x, tid = threadIdx.x;
    if (tid < NW * 2) {
        float acc = b2[tid];
        for (int i = 0; i < CC; ++i) acc += h[b * CC + i] * w2[i * (NW * 2) + tid];
        int nw = tid >> 1;
        if ((tid & 1) == 0) {
            float sp = (acc > 20.0f) ? acc : log1pf(__expf(acc));   // softplus
            s_a[nw] = sp + 0.01f;
        } else {
            s_b[nw] = (float)TT * fast_sigmoid(acc);                // sigmoid * SEQ_LEN
        }
    }
    __syncthreads();
    for (int t = tid; t < TT; t += 256) {
        #pragma unroll
        for (int w = 0; w < NW; ++w) {
            float z  = ((float)t - s_b[w]) * fast_rcp(s_a[w]);
            float z2 = z * z;
            psi[((size_t)b * TT + t) * NW + w] = (1.0f - z2) * __expf(-0.5f * z2);
        }
    }
}

// ============================================================
// WMMA GEMM: C[M,512] = A[M,K] @ B[K,512], bf16 x bf16 -> f32
//   MODE 0: A = Cheby basis of tanh(trend_ctx), built in staging (K=2048)
//   MODE 1: A = psi[row,w] * spike_ctx[row,c], k = w*512+c     (K=2048)
//   MODE 2: A = bf16 matrix, plain copy                        (K=512)
// Block: 256 threads (8 wave32), 128x128 tile, BK=32, double-buffered.
// A tile: VALU-computed, written fragment-permuted with ds_store_b16.
// B tile: weights pre-permuted in HBM -> one contiguous 8KB block per
//         K-step, moved with GLOBAL_LOAD_ASYNC_TO_LDS_B128 (ASYNCcnt).
// ============================================================
template <int MODE>
__device__ __forceinline__ void stage_A(unsigned short (*sA)[32][16],
                                        const unsigned short* __restrict__ Abase,
                                        const float* __restrict__ Psi,
                                        int arow, int ar, int ah, int kt, int Ktot) {
    float vals[16];
    const int kg0 = kt + ah * 16;
    if (MODE == 0) {
        // Chebyshev basis on the fly: 16 k's cover 4 channels x 4 degrees.
        // 4 bf16 in one 64-bit load; branch-free tanh on v_exp_f32/v_rcp_f32.
        int c0 = kg0 >> 2;
        uint2 tp = *(const uint2*)(Abase + (size_t)arow * CC + c0);
        float xt[4];
        xt[0] = fast_tanh(bf2f((unsigned short)(tp.x & 0xffffu)));
        xt[1] = fast_tanh(bf2f((unsigned short)(tp.x >> 16)));
        xt[2] = fast_tanh(bf2f((unsigned short)(tp.y & 0xffffu)));
        xt[3] = fast_tanh(bf2f((unsigned short)(tp.y >> 16)));
        #pragma unroll
        for (int j = 0; j < 16; ++j) {
            float v = xt[j >> 2], v2 = v * v;
            int d = j & 3;
            vals[j] = (d == 0) ? 1.0f
                    : (d == 1) ? v
                    : (d == 2) ? (2.0f * v2 - 1.0f)
                               : v * (4.0f * v2 - 3.0f);
        }
    } else if (MODE == 1) {
        int w  = kg0 >> 9;
        int c0 = kg0 & 511;
        float p = Psi[(size_t)arow * NW + w];
        load16bf(Abase + (size_t)arow * CC + c0, vals);
        #pragma unroll
        for (int j = 0; j < 16; ++j) vals[j] *= p;
    } else {
        load16bf(Abase + (size_t)arow * Ktot + kg0, vals);
    }
    const int mtile = ar >> 4, m = ar & 15;
    #pragma unroll
    for (int j = 0; j < 16; ++j) {
        int kk   = ah * 16 + j;
        int half = kk >> 4;
        int e    = kk & 15;
        sA[mtile][half * 16 + m][e] = f2bf(vals[j]);
    }
}

__device__ __forceinline__ void stage_B_async(void* ldsDst,
                                              const unsigned short* __restrict__ Wp,
                                              int colBase, int kt, int tid) {
    // 8KB contiguous fragment-ordered block: kTile row, 8 nTiles starting
    // at colBase/16 (32 nTiles per kTile, so the 8 we need are contiguous).
    const unsigned short* bsrc =
        Wp + ((size_t)(kt >> 5) * 32 + (colBase >> 4)) * 512;
    unsigned long long ga = (unsigned long long)(uintptr_t)bsrc + (unsigned)(tid * 32);
    unsigned lds0 = (unsigned)(uintptr_t)ldsDst + (unsigned)(tid * 32);
    asm volatile("global_load_async_to_lds_b128 %0, %1, off"
                 :: "v"(lds0), "v"(ga) : "memory");
    asm volatile("global_load_async_to_lds_b128 %0, %1, off offset:16"
                 :: "v"(lds0), "v"(ga) : "memory");
}

template <int MODE, int KTOT>
__global__ __launch_bounds__(256)
void jdkan_wmma_gemm_kernel(const unsigned short* __restrict__ Abase,
                            const unsigned short* __restrict__ Wp,
                            const float* __restrict__ Psi,
                            float* __restrict__ Cout,
                            unsigned short* __restrict__ CoutBf) {
    __shared__ __attribute__((aligned(32))) unsigned short sA[2][8][32][16]; // 2 x 8 KB
    __shared__ __attribute__((aligned(32))) unsigned short sB[2][8][32][16]; // 2 x 8 KB

    const int tid     = threadIdx.x;
    const int rowBase = blockIdx.x * 128;
    const int colBase = blockIdx.y * 128;

    // A staging coords: one row-half (16 k's) per thread
    const int ar   = tid >> 1;         // 0..127 row within tile
    const int ah   = tid & 1;          // which 16-wide k half
    const int arow = rowBase + ar;

    const int wave = tid >> 5, lane = tid & 31;
    const int wr = wave >> 1, wc = wave & 1;
    const int mt0 = wr * 2;            // first m-tile owned by this wave
    const int nt0 = wc * 4;            // first n-tile owned by this wave

    v8f acc[2][4] = {};

    // prologue: fill buffer 0
    stage_A<MODE>(sA[0], Abase, Psi, arow, ar, ah, 0, KTOT);
    stage_B_async(&sB[0][0][0][0], Wp, colBase, 0, tid);
    asm volatile("s_wait_asynccnt 0x0" ::: "memory");
    __syncthreads();

    for (int kt = 0; kt < KTOT; kt += 32) {
        const int cur = (kt >> 5) & 1;
        const int nxt = cur ^ 1;

        // prefetch next K-step into the other buffer while we compute
        if (kt + 32 < KTOT) {
            stage_A<MODE>(sA[nxt], Abase, Psi, arow, ar, ah, kt + 32, KTOT);
            stage_B_async(&sB[nxt][0][0][0], Wp, colBase, kt + 32, tid);
        }

        v16bf afrag[2], bfrag[4];
        #pragma unroll
        for (int mt = 0; mt < 2; ++mt)
            afrag[mt] = *(const v16bf*)&sA[cur][mt0 + mt][lane][0];
        #pragma unroll
        for (int nt = 0; nt < 4; ++nt)
            bfrag[nt] = *(const v16bf*)&sB[cur][nt0 + nt][lane][0];
        #pragma unroll
        for (int mt = 0; mt < 2; ++mt)
            #pragma unroll
            for (int nt = 0; nt < 4; ++nt)
                acc[mt][nt] = __builtin_amdgcn_wmma_f32_16x16x32_bf16(
                    false, afrag[mt], false, bfrag[nt],
                    (short)0, acc[mt][nt], false, false);

        asm volatile("s_wait_asynccnt 0x0" ::: "memory");
        __syncthreads();
    }

    // ---------------- epilogue ----------------
    const int n = lane & 15;
    const int halfSel = lane >> 4;
    #pragma unroll
    for (int mt = 0; mt < 2; ++mt) {
        int rowB = rowBase + wr * 32 + mt * 16;
        #pragma unroll
        for (int nt = 0; nt < 4; ++nt) {
            int colB = colBase + wc * 64 + nt * 16 + n;
            #pragma unroll
            for (int v = 0; v < 8; ++v) {
                int m = v + halfSel * 8;
                float val = acc[mt][nt][v];
                size_t idx = (size_t)(rowB + m) * CC + colB;
                Cout[idx] = val;
                if (MODE == 0) CoutBf[idx] = f2bf(val);   // bf16 copy for amp GEMM
            }
        }
    }
}

// ============================================================
// Fused amp-gate + residual + LayerNorm. One block per row.
// ============================================================
__global__ void final_ln_kernel(const float* __restrict__ trend_out,
                                const float* __restrict__ spike_out,
                                const float* __restrict__ amp_lin,
                                const float* __restrict__ x,
                                const float* __restrict__ amp_b,
                                const float* __restrict__ gamma,
                                const float* __restrict__ beta,
                                float* __restrict__ out) {
    __shared__ float red[256];
    const int row = blockIdx.x;
    const int tid = threadIdx.x;
    const size_t rb = (size_t)row * CC;

    float y[2];
    float s = 0.0f;
    #pragma unroll
    for (int i = 0; i < 2; ++i) {
        int c = tid + i * 256;
        float amp = 2.0f * fast_sigmoid(amp_lin[rb + c] + amp_b[c]);
        float v   = trend_out[rb + c] + spike_out[rb + c] * amp + x[rb + c];
        y[i] = v; s += v;
    }
    red[tid] = s; __syncthreads();
    for (int st = 128; st > 0; st >>= 1) { if (tid < st) red[tid] += red[tid + st]; __syncthreads(); }
    float mu = red[0] * (1.0f / (float)CC);
    __syncthreads();
    float s2 = 0.0f;
    #pragma unroll
    for (int i = 0; i < 2; ++i) { float d = y[i] - mu; s2 += d * d; }
    red[tid] = s2; __syncthreads();
    for (int st = 128; st > 0; st >>= 1) { if (tid < st) red[tid] += red[tid + st]; __syncthreads(); }
    float rstd = rsqrtf(red[0] * (1.0f / (float)CC) + 1e-5f);
    #pragma unroll
    for (int i = 0; i < 2; ++i) {
        int c = tid + i * 256;
        out[rb + c] = (y[i] - mu) * rstd * gamma[c] + beta[c];
    }
}

// ============================================================
// Host launcher
// ============================================================
extern "C" void kernel_launch(void* const* d_in, const int* in_sizes, int n_in,
                              void* d_out, int out_size, void* d_ws, size_t ws_size,
                              hipStream_t stream) {
    const float* x        = (const float*)d_in[0];
    const float* tconv_w  = (const float*)d_in[1];
    const float* tconv_b  = (const float*)d_in[2];
    const float* sconv_w  = (const float*)d_in[3];
    const float* sconv_b  = (const float*)d_in[4];
    const float* lconv_w  = (const float*)d_in[5];
    const float* lconv_b  = (const float*)d_in[6];
    const float* cheby    = (const float*)d_in[7];
    const float* hyper_w1 = (const float*)d_in[8];
    const float* hyper_b1 = (const float*)d_in[9];
    const float* hyper_w2 = (const float*)d_in[10];
    const float* hyper_b2 = (const float*)d_in[11];
    const float* amp_w    = (const float*)d_in[12];
    const float* amp_b    = (const float*)d_in[13];
    const float* wavelet  = (const float*)d_in[14];
    const float* ln_gamma = (const float*)d_in[15];
    const float* ln_beta  = (const float*)d_in[16];
    float* out = (float*)d_out;

    // ---- workspace carve-up ----
    char* ws = (char*)d_ws;
    size_t off = 0;
    auto carve = [&](size_t bytes) -> char* {
        char* p = ws + off;
        off += (bytes + 255) & ~(size_t)255;
        return p;
    };
    unsigned short* trend_ctx_bf = (unsigned short*)carve((size_t)MROWS * CC * 2); // 16 MB
    unsigned short* spike_ctx_bf = (unsigned short*)carve((size_t)MROWS * CC * 2); // 16 MB
    float*          trend_out    = (float*)         carve((size_t)MROWS * CC * 4); // 32 MB
    unsigned short* trend_out_bf = (unsigned short*)carve((size_t)MROWS * CC * 2); // 16 MB
    float*          spike_out    = (float*)         carve((size_t)MROWS * CC * 4); // 32 MB
    float*          amp_lin      = (float*)         carve((size_t)MROWS * CC * 4); // 32 MB
    unsigned short* Wc           = (unsigned short*)carve((size_t)KCHEB * CC * 2); //  2 MB
    unsigned short* Wv           = (unsigned short*)carve((size_t)KWAVE * CC * 2); //  2 MB
    unsigned short* Wa           = (unsigned short*)carve((size_t)CC * CC * 2);    // .5 MB
    float*          slope_pool   = (float*)         carve((size_t)BB * CC * 4);
    float*          value_pool   = (float*)         carve((size_t)BB * CC * 4);
    float*          h_buf        = (float*)         carve((size_t)BB * CC * 4);
    float*          psi          = (float*)         carve((size_t)BB * TT * NW * 4);
    (void)ws_size; (void)in_sizes; (void)n_in; (void)out_size;

    // ---- 1. pack weights to bf16 (fragment-permuted) ----
    pack_weight_kernel<0><<<(KCHEB * CC) / 256, 256, 0, stream>>>(cheby,   Wc);
    pack_weight_kernel<1><<<(KWAVE * CC) / 256, 256, 0, stream>>>(wavelet, Wv);
    pack_weight_kernel<2><<<(CC * CC)    / 256, 256, 0, stream>>>(amp_w,   Wa);

    // ---- 2. decomposition scan ----
    decomp_scan_kernel<<<(BB * CC) / 256, 256, 0, stream>>>(
        x, tconv_w, tconv_b, sconv_w, sconv_b, lconv_w, lconv_b,
        trend_ctx_bf, spike_ctx_bf, slope_pool, value_pool);

    // ---- 3. hypernet ----
    hyper1_kernel<<<(BB * CC) / 256, 256, 0, stream>>>(
        slope_pool, value_pool, hyper_w1, hyper_b1, h_buf);
    hyper2_psi_kernel<<<BB, 256, 0, stream>>>(h_buf, hyper_w2, hyper_b2, psi);

    // ---- 4. WMMA GEMMs ----
    dim3 ggrid(MROWS / 128, CC / 128);
    // trend_out = ChebyBasis(trend_ctx) @ Wc   (also emit bf16 copy for amp GEMM)
    jdkan_wmma_gemm_kernel<0, KCHEB><<<ggrid, 256, 0, stream>>>(
        trend_ctx_bf, Wc, nullptr, trend_out, trend_out_bf);
    // spike_out = (psi * spike_ctx) @ Wv
    jdkan_wmma_gemm_kernel<1, KWAVE><<<ggrid, 256, 0, stream>>>(
        spike_ctx_bf, Wv, psi, spike_out, nullptr);
    // amp_lin = trend_out @ amp_w
    jdkan_wmma_gemm_kernel<2, CC><<<ggrid, 256, 0, stream>>>(
        trend_out_bf, Wa, nullptr, amp_lin, nullptr);

    // ---- 5. gate + residual + LayerNorm ----
    final_ln_kernel<<<MROWS, 256, 0, stream>>>(
        trend_out, spike_out, amp_lin, x, amp_b, ln_gamma, ln_beta, out);
}
